// GAT_87282325389548
// MI455X (gfx1250) — matrix operand order repaired and checked by reading, not compiled, not run on hardware
//
#include <hip/hip_runtime.h>
#include <hip/hip_bf16.h>
#include <cstring>

typedef __bf16 bf16;
typedef __attribute__((ext_vector_type(16))) __bf16 v16bf;
typedef __attribute__((ext_vector_type(8)))  __bf16 v8bf;
typedef __attribute__((ext_vector_type(8)))  float  v8f;

union frag_u { v16bf v; v8bf h[2]; };

// ---------------- helpers ----------------

__device__ inline unsigned f2mono(float f) {
  unsigned u = __float_as_uint(f);
  return (u & 0x80000000u) ? ~u : (u | 0x80000000u);
}
__device__ inline float mono2f(unsigned u) {
  return (u & 0x80000000u) ? __uint_as_float(u & 0x7FFFFFFFu) : __uint_as_float(~u);
}
static unsigned host_f2mono(float f) {
  unsigned u; memcpy(&u, &f, 4);
  return (u & 0x80000000u) ? ~u : (u | 0x80000000u);
}

// ---------------- elementwise utility kernels ----------------

__global__ void k_f32_to_bf16(const float* __restrict__ x, bf16* __restrict__ y, size_t n) {
  size_t i = blockIdx.x * (size_t)blockDim.x + threadIdx.x;
  if (i < n) y[i] = (bf16)x[i];
}

__global__ void k_fill_f32(float* __restrict__ p, float v, size_t n) {
  size_t i = blockIdx.x * (size_t)blockDim.x + threadIdx.x;
  if (i < n) p[i] = v;
}

__global__ void k_fill_u32(unsigned* __restrict__ p, unsigned v, size_t n) {
  size_t i = blockIdx.x * (size_t)blockDim.x + threadIdx.x;
  if (i < n) p[i] = v;
}

__global__ void k_relu_bias_to_bf16(const float* __restrict__ x, const float* __restrict__ bias,
                                    bf16* __restrict__ y, size_t n, int cols) {
  size_t i = blockIdx.x * (size_t)blockDim.x + threadIdx.x;
  if (i >= n) return;
  float v = x[i] + bias[i % cols];
  y[i] = (bf16)(v > 0.f ? v : 0.f);
}

// Pre-pack weight matrix W[K,N] (f32 row-major) into per-lane WMMA B-fragment layout:
// Bp[((kt*(N/16) + nt)*32 + lane)*16 + e] = bf16(W[(kt*32 + (e>>3)*16 + (lane>>4)*8 + (e&7))*N + nt*16 + (lane&15)])
// so a B fragment is 32 contiguous bytes per lane (two b128 loads, zero packing movs).
__global__ void k_pack_B(const float* __restrict__ W, bf16* __restrict__ Bp, int K, int N) {
  size_t i = blockIdx.x * (size_t)blockDim.x + threadIdx.x;
  size_t total = (size_t)K * N;
  if (i >= total) return;
  int e    = (int)(i & 15);
  int lane = (int)((i >> 4) & 31);
  size_t tile = i >> 9;               // / 512
  int tilesN16 = N >> 4;
  int kt = (int)(tile / tilesN16);
  int nt = (int)(tile % tilesN16);
  int k = kt * 32 + ((e >> 3) << 4) + ((lane >> 4) << 3) + (e & 7);
  int n = (nt << 4) + (lane & 15);
  Bp[i] = (bf16)W[(size_t)k * N + n];
}

// ---------------- WMMA bf16 GEMM (D = A[MxK] * Bpacked, f32 accum) ----------------
// Fragment layouts per CDNA5 ISA 7.12.2 (wave32).

__device__ inline v16bf load_a_frag(const bf16* __restrict__ A, int lda, int mBase,
                                    int kBase, int lane, int Mlast) {
  int m = mBase + (lane & 15);
  m = m < Mlast ? m : Mlast;          // clamp: padding rows compute garbage, discarded at store
  const bf16* p = A + (size_t)m * lda + kBase + (lane >> 4) * 8;
  frag_u u;
  u.h[0] = *(const v8bf*)(p);         // K = kBase + half*8 + [0..7]
  u.h[1] = *(const v8bf*)(p + 16);    // K = kBase + 16 + half*8 + [0..7]
  return u.v;
}

__device__ inline v16bf load_b_packed(const bf16* __restrict__ Bp, int tileIdx, int lane) {
  const bf16* p = Bp + (((size_t)tileIdx * 32 + lane) << 4);
  frag_u u;
  u.h[0] = *(const v8bf*)(p);
  u.h[1] = *(const v8bf*)(p + 8);
  return u.v;
}

__device__ inline void store_d_frag(float* __restrict__ D, int ldd, int mBase, int nBase,
                                    int lane, int M, v8f c) {
  int n = nBase + (lane & 15);
  int m0 = mBase + ((lane >> 4) << 3);   // VGPR r -> row m0 + r
  float* p = D + (size_t)m0 * ldd + n;
  if (m0 + 7 < M) {                      // fast path: whole fragment row-range valid
#pragma unroll
    for (int r = 0; r < 8; ++r) p[(size_t)r * ldd] = c[r];
  } else {
#pragma unroll
    for (int r = 0; r < 8; ++r) if (m0 + r < M) p[(size_t)r * ldd] = c[r];
  }
}

__global__ __launch_bounds__(256) void k_gemm_bf16(
    const bf16* __restrict__ A, const bf16* __restrict__ Bp, float* __restrict__ D,
    int M, int N, int K) {
  int lane = threadIdx.x & 31;
  int wid  = blockIdx.x * (blockDim.x >> 5) + (threadIdx.x >> 5);
  int tilesN = N >> 5;
  int tilesM = (M + 31) >> 5;
  if (wid >= tilesM * tilesN) return;
  int tm = (wid / tilesN) << 5;
  int tn = (wid % tilesN) << 5;
  int tilesN16 = N >> 4;
  int nt0 = tn >> 4;

  v8f acc00 = {}, acc01 = {}, acc10 = {}, acc11 = {};
  for (int k0 = 0; k0 < K; k0 += 32) {
    int kt = k0 >> 5;
    v16bf a0 = load_a_frag(A, K, tm,      k0, lane, M - 1);
    v16bf a1 = load_a_frag(A, K, tm + 16, k0, lane, M - 1);
    v16bf b0 = load_b_packed(Bp, kt * tilesN16 + nt0,     lane);
    v16bf b1 = load_b_packed(Bp, kt * tilesN16 + nt0 + 1, lane);
    if (k0 + 32 < K) {
      // pull the next A K-slice toward the WGP while WMMAs run
      __builtin_prefetch(A + (size_t)(tm + (lane & 15)) * K + k0 + 32, 0, 1);
    }
    acc00 = __builtin_amdgcn_wmma_f32_16x16x32_bf16(false, a0, false, b0, (short)0, acc00, false, false);
    acc01 = __builtin_amdgcn_wmma_f32_16x16x32_bf16(false, a0, false, b1, (short)0, acc01, false, false);
    acc10 = __builtin_amdgcn_wmma_f32_16x16x32_bf16(false, a1, false, b0, (short)0, acc10, false, false);
    acc11 = __builtin_amdgcn_wmma_f32_16x16x32_bf16(false, a1, false, b1, (short)0, acc11, false, false);
  }
  store_d_frag(D, N, tm,      tn,      lane, M, acc00);
  store_d_frag(D, N, tm,      tn + 16, lane, M, acc01);
  store_d_frag(D, N, tm + 16, tn,      lane, M, acc10);
  store_d_frag(D, N, tm + 16, tn + 16, lane, M, acc11);
}

// ---------------- attention coefficient projections ----------------
// One wave per node: alpha_src[n][h] = sum_c h[n,h,c]*a_src[h,c]; same for dst.

template<int H, int C>
__global__ __launch_bounds__(256) void k_alpha(
    const float* __restrict__ h, const float* __restrict__ a_src,
    const float* __restrict__ a_dst, float* __restrict__ as, float* __restrict__ ad,
    long long N) {
  constexpr int HC = H * C;
  constexpr int PL = HC / 32;   // elements per lane
  constexpr int LPH = C / PL;   // lanes per head (power of 2, lane-aligned)
  int lane = threadIdx.x & 31;
  long long n = blockIdx.x * (long long)(blockDim.x >> 5) + (threadIdx.x >> 5);
  if (n >= N) return;
  int col = lane * PL;
  int head = col / C;
  const float* row = h + (size_t)n * HC + col;
  const float* sv = a_src + head * C + (col % C);
  const float* dv = a_dst + head * C + (col % C);
  float s = 0.f, d = 0.f;
#pragma unroll
  for (int i = 0; i < PL; ++i) { float v = row[i]; s += v * sv[i]; d += v * dv[i]; }
#pragma unroll
  for (int off = LPH >> 1; off > 0; off >>= 1) {
    s += __shfl_xor(s, off, 32);
    d += __shfl_xor(d, off, 32);
  }
  if ((lane & (LPH - 1)) == 0) {
    as[(size_t)n * H + head] = s;
    ad[(size_t)n * H + head] = d;
  }
}

// ---------------- edge softmax (3 passes) ----------------

template<int H>
__global__ void k_edge_e(const float* __restrict__ as, const float* __restrict__ ad,
                         const int* __restrict__ ei, long long E, long long Ep,
                         float* __restrict__ ebuf, unsigned* __restrict__ mmono) {
  long long idx = blockIdx.x * (long long)blockDim.x + threadIdx.x;
  if (idx >= Ep * H) return;
  long long e = idx / H;
  int hh = (int)(idx % H);
  int s, d;
  if (e < E) { s = ei[e]; d = ei[E + e]; } else { s = d = (int)(e - E); }
  float v = as[(size_t)s * H + hh] + ad[(size_t)d * H + hh];
  v = (v >= 0.f) ? v : 0.2f * v;               // LeakyReLU(0.2)
  ebuf[idx] = v;
  atomicMax(&mmono[(size_t)d * H + hh], f2mono(v));
}

template<int H>
__global__ void k_edge_p(float* __restrict__ ebuf, const unsigned* __restrict__ mmono,
                         float* __restrict__ den, const int* __restrict__ ei,
                         long long E, long long Ep) {
  long long idx = blockIdx.x * (long long)blockDim.x + threadIdx.x;
  if (idx >= Ep * H) return;
  long long e = idx / H;
  int hh = (int)(idx % H);
  int d;
  if (e < E) d = ei[E + e]; else d = (int)(e - E);
  float m = mono2f(mmono[(size_t)d * H + hh]);
  float p = __expf(ebuf[idx] - m);
  ebuf[idx] = p;
  atomicAdd(&den[(size_t)d * H + hh], p);
}

// One wave per edge: out[dst] += h[src] * alpha  (L2-resident gather/scatter)
template<int H, int C>
__global__ __launch_bounds__(256) void k_edge_scatter(
    const float* __restrict__ hfeat, const float* __restrict__ ebuf,
    const float* __restrict__ den, const int* __restrict__ ei,
    long long E, long long Ep, float* __restrict__ out) {
  constexpr int HC = H * C;
  constexpr int PL = HC / 32;
  int lane = threadIdx.x & 31;
  long long e = blockIdx.x * (long long)(blockDim.x >> 5) + (threadIdx.x >> 5);
  if (e >= Ep) return;
  int s, d;
  if (e < E) { s = ei[e]; d = ei[E + e]; } else { s = d = (int)(e - E); }
  int col = lane * PL;
  int head = col / C;
  float alpha = ebuf[(size_t)e * H + head] / (den[(size_t)d * H + head] + 1e-16f);
  const float* hs = hfeat + (size_t)s * HC + col;
  float* od = out + (size_t)d * HC + col;
#pragma unroll
  for (int j = 0; j < PL; ++j) atomicAdd(&od[j], hs[j] * alpha);
}

// ---------------- bias + log_softmax over 64 columns (one wave per row) ----------------

__global__ __launch_bounds__(256) void k_bias_logsoftmax64(
    const float* __restrict__ x, const float* __restrict__ bias,
    float* __restrict__ out, long long N) {
  int lane = threadIdx.x & 31;
  long long n = blockIdx.x * (long long)(blockDim.x >> 5) + (threadIdx.x >> 5);
  if (n >= N) return;
  const float* row = x + (size_t)n * 64;
  float v0 = row[lane * 2 + 0] + bias[lane * 2 + 0];
  float v1 = row[lane * 2 + 1] + bias[lane * 2 + 1];
  float m = fmaxf(v0, v1);
#pragma unroll
  for (int off = 16; off > 0; off >>= 1) m = fmaxf(m, __shfl_xor(m, off, 32));
  float s = __expf(v0 - m) + __expf(v1 - m);
#pragma unroll
  for (int off = 16; off > 0; off >>= 1) s += __shfl_xor(s, off, 32);
  float l = m + __logf(s);
  float* orow = out + (size_t)n * 64;
  orow[lane * 2 + 0] = v0 - l;
  orow[lane * 2 + 1] = v1 - l;
}

// ---------------- launcher ----------------

extern "C" void kernel_launch(void* const* d_in, const int* in_sizes, int n_in,
                              void* d_out, int out_size, void* d_ws, size_t ws_size,
                              hipStream_t stream) {
  (void)n_in; (void)out_size; (void)ws_size;
  const float* x      = (const float*)d_in[0];
  const int*   ei     = (const int*)d_in[1];
  const float* W1     = (const float*)d_in[2];
  const float* a_src1 = (const float*)d_in[3];
  const float* a_dst1 = (const float*)d_in[4];
  const float* b1     = (const float*)d_in[5];
  const float* W2     = (const float*)d_in[6];
  const float* a_src2 = (const float*)d_in[7];
  const float* a_dst2 = (const float*)d_in[8];
  const float* b2     = (const float*)d_in[9];

  const long long N  = in_sizes[0] / 128; // Fin = 128
  const long long E  = in_sizes[1] / 2;
  const long long Ep = E + N;             // with self-loops

  // workspace carve (256B aligned)
  char* w = (char*)d_ws;
  auto carve = [&](size_t bytes) -> void* {
    void* p = (void*)w;
    w += (bytes + 255) & ~(size_t)255;
    return p;
  };
  bf16*     x_bf  = (bf16*)carve((size_t)N * 128 * 2);
  bf16*     W1_bp = (bf16*)carve((size_t)128 * 256 * 2);  // packed B fragments
  bf16*     W2_bp = (bf16*)carve((size_t)256 * 64 * 2);   // packed B fragments
  float*    h1    = (float*)carve((size_t)N * 256 * 4);
  float*    as1   = (float*)carve((size_t)N * 4 * 4);
  float*    ad1   = (float*)carve((size_t)N * 4 * 4);
  unsigned* mm1   = (unsigned*)carve((size_t)N * 4 * 4);
  float*    den1  = (float*)carve((size_t)N * 4 * 4);
  float*    eb1   = (float*)carve((size_t)Ep * 4 * 4);
  float*    o1    = (float*)carve((size_t)N * 256 * 4);
  bf16*     h1b   = (bf16*)carve((size_t)N * 256 * 2);
  float*    h2    = (float*)carve((size_t)N * 64 * 4);
  float*    as2   = (float*)carve((size_t)N * 4);
  float*    ad2   = (float*)carve((size_t)N * 4);
  unsigned* mm2   = (unsigned*)carve((size_t)N * 4);
  float*    den2  = (float*)carve((size_t)N * 4);
  float*    eb2   = (float*)carve((size_t)Ep * 4);
  float*    o2    = (float*)carve((size_t)N * 64 * 4);

  const unsigned NEG_INF_MONO = host_f2mono(-3.0e38f);
  auto blk = [](long long work, int tpb) { return (unsigned)((work + tpb - 1) / tpb); };

  // --- precision conversion + weight packing ---
  k_f32_to_bf16<<<blk(N * 128, 256), 256, 0, stream>>>(x, x_bf, (size_t)N * 128);
  k_pack_B<<<blk(128 * 256, 256), 256, 0, stream>>>(W1, W1_bp, 128, 256);
  k_pack_B<<<blk(256 * 64, 256), 256, 0, stream>>>(W2, W2_bp, 256, 64);

  // --- layer 1: h1 = x @ W1 (bf16 WMMA, f32 accum) ---
  {
    int tilesM = (int)((N + 31) / 32), tilesN = 256 / 32;
    long long waves = (long long)tilesM * tilesN;
    k_gemm_bf16<<<blk(waves, 8), 256, 0, stream>>>(x_bf, W1_bp, h1, (int)N, 256, 128);
  }
  k_alpha<4, 64><<<blk(N, 8), 256, 0, stream>>>(h1, a_src1, a_dst1, as1, ad1, N);

  k_fill_u32<<<blk(N * 4, 256), 256, 0, stream>>>(mm1, NEG_INF_MONO, (size_t)N * 4);
  k_fill_f32<<<blk(N * 4, 256), 256, 0, stream>>>(den1, 0.f, (size_t)N * 4);
  k_fill_f32<<<blk(N * 256, 256), 256, 0, stream>>>(o1, 0.f, (size_t)N * 256);

  k_edge_e<4><<<blk(Ep * 4, 256), 256, 0, stream>>>(as1, ad1, ei, E, Ep, eb1, mm1);
  k_edge_p<4><<<blk(Ep * 4, 256), 256, 0, stream>>>(eb1, mm1, den1, ei, E, Ep);
  k_edge_scatter<4, 64><<<blk(Ep, 8), 256, 0, stream>>>(h1, eb1, den1, ei, E, Ep, o1);

  // --- ReLU(o1 + b1) -> bf16 input for layer 2 ---
  k_relu_bias_to_bf16<<<blk(N * 256, 256), 256, 0, stream>>>(o1, b1, h1b, (size_t)N * 256, 256);

  // --- layer 2: h2 = h1b @ W2 ---
  {
    int tilesM = (int)((N + 31) / 32), tilesN = 64 / 32;
    long long waves = (long long)tilesM * tilesN;
    k_gemm_bf16<<<blk(waves, 8), 256, 0, stream>>>(h1b, W2_bp, h2, (int)N, 64, 256);
  }
  k_alpha<1, 64><<<blk(N, 8), 256, 0, stream>>>(h2, a_src2, a_dst2, as2, ad2, N);

  k_fill_u32<<<blk(N, 256), 256, 0, stream>>>(mm2, NEG_INF_MONO, (size_t)N);
  k_fill_f32<<<blk(N, 256), 256, 0, stream>>>(den2, 0.f, (size_t)N);
  k_fill_f32<<<blk(N * 64, 256), 256, 0, stream>>>(o2, 0.f, (size_t)N * 64);

  k_edge_e<1><<<blk(Ep, 256), 256, 0, stream>>>(as2, ad2, ei, E, Ep, eb2, mm2);
  k_edge_p<1><<<blk(Ep, 256), 256, 0, stream>>>(eb2, mm2, den2, ei, E, Ep);
  k_edge_scatter<1, 64><<<blk(Ep, 8), 256, 0, stream>>>(h2, eb2, den2, ei, E, Ep, o2);

  // --- bias + log_softmax -> d_out ---
  k_bias_logsoftmax64<<<blk(N, 8), 256, 0, stream>>>(o2, b2, (float*)d_out, N);
}
